// RNN_70317204570606
// MI455X (gfx1250) — matrix operand order repaired and checked by reading, not compiled, not run on hardware
//
#include <hip/hip_runtime.h>

// ---------------------------------------------------------------------------
// RNN forward for MI455X (gfx1250, wave32, WMMA).
//   T=512, B=64, D_IN=512, D_H=1024, D_OUT=512
//   k_pack    : fp32 weights -> bf16, pre-swizzled into WMMA-B tile layout
//   k_gemm_xh : xh = X @ W_xh + b_h      (bf16 WMMA, fp32 accum, 4 N-tiles/wave)
//   k_init    : h_buf0 = hidden ; barrier ctr = 0
//   k_scan    : persistent, 512 steps, h = tanh(xh_t + h @ W_hh)
//               A staged in LDS (WMMA-A layout), one grid barrier per step
//   k_gemm_out: out = hs @ W_hq + b_q    (fp32 to d_out, 4 N-tiles/wave)
//   k_final   : copy final hidden to tail of d_out
// ---------------------------------------------------------------------------

typedef __attribute__((ext_vector_type(16))) __bf16 v16bf;
typedef __attribute__((ext_vector_type(8)))  float  v8f;

#define T_STEPS 512
#define BATCH   64
#define D_IN    512
#define D_H     1024
#define D_OUT   512

// ---- bf16 helpers: use native converts (v_cvt_*bf16*) instead of bit-twiddle
__device__ __forceinline__ unsigned short f2bfu(float x) {
    union { __bf16 b; unsigned short u; } o; o.b = (__bf16)x; return o.u;
}
__device__ __forceinline__ float bfu2f(unsigned short s) {
    union { unsigned u; float f; } v; v.u = ((unsigned)s) << 16;
    return v.f;
}
__device__ __forceinline__ unsigned pk2(float lo, float hi) {
    union { __bf16 b[2]; unsigned u; } o;
    o.b[0] = (__bf16)lo; o.b[1] = (__bf16)hi;
    return o.u;
}

// A-operand K index for (lane, element): 16-bit A 16x32 layout (ISA 7.12.2):
//   K(e) = (e&7) + 8*hi + 16*(e>>3)  ->  two contiguous runs of 8 per lane.
__device__ __forceinline__ int kamap(int hi, int e) {
    return (e & 7) + (hi << 3) + ((e >> 3) << 4);
}

// Build one WMMA-A fragment (row-major fp32 source) with 4 x b128 loads.
__device__ __forceinline__ v16bf load_a_f32(const float* __restrict__ rowp, int hi) {
    const float4* p0 = (const float4*)(rowp + hi * 8);        // K = hi*8 .. +7
    const float4* p1 = (const float4*)(rowp + hi * 8 + 16);   // K = hi*8+16 .. +7
    float4 x0 = p0[0], x1 = p0[1], y0 = p1[0], y1 = p1[1];
    union { v16bf v; unsigned u[8]; } o;
    o.u[0] = pk2(x0.x, x0.y); o.u[1] = pk2(x0.z, x0.w);
    o.u[2] = pk2(x1.x, x1.y); o.u[3] = pk2(x1.z, x1.w);
    o.u[4] = pk2(y0.x, y0.y); o.u[5] = pk2(y0.z, y0.w);
    o.u[6] = pk2(y1.x, y1.y); o.u[7] = pk2(y1.z, y1.w);
    return o.v;
}

// Build one WMMA-A fragment (row-major bf16 source) with 2 x b128 loads.
__device__ __forceinline__ v16bf load_a_bf16(const unsigned short* __restrict__ rowp, int hi) {
    union { v16bf v; uint4 q[2]; } o;
    o.q[0] = *(const uint4*)(rowp + hi * 8);
    o.q[1] = *(const uint4*)(rowp + hi * 8 + 16);
    return o.v;
}

// ---------------------------------------------------------------------------
// Pack fp32 W[K][N] row-major into bf16 WMMA-B tiles:
//   P[(kt*Nt + nt)*512 + lane*16 + e] = W[kt*32 + (e&7) + 8*(lane>=16) + 16*(e>>3)][nt*16 + (lane&15)]
// ---------------------------------------------------------------------------
__global__ void k_pack(const float* __restrict__ W, unsigned short* __restrict__ P, int N) {
    int idx  = blockIdx.x * 256 + threadIdx.x;
    int e    = idx & 15;
    int lane = (idx >> 4) & 31;
    int tile = idx >> 9;
    int Nt   = N >> 4;
    int nt   = tile % Nt;
    int kt   = tile / Nt;
    int k    = kt * 32 + (e & 7) + ((lane >> 4) << 3) + ((e >> 3) << 4);
    int n    = nt * 16 + (lane & 15);
    P[idx]   = f2bfu(W[(size_t)k * N + n]);
}

// ---------------------------------------------------------------------------
// xh = X @ W_xh + b_h -> bf16 [T*B, D_H].  4 N-tiles per wave (A reused 4x).
// ---------------------------------------------------------------------------
__global__ void k_gemm_xh(const float* __restrict__ X,
                          const unsigned short* __restrict__ Wp,
                          const float* __restrict__ bh,
                          unsigned short* __restrict__ xh) {
    const int tid  = threadIdx.x;
    const int lane = tid & 31, wave = tid >> 5;
    const int r = lane & 15, hi = lane >> 4;
    const int job = blockIdx.x * 8 + wave;             // 2048 * 16 jobs
    const int mt  = job >> 4;                          // 0..2047
    const int ng  = job & 15;                          // group of 4 N-tiles
    const int row = mt * 16 + r;

    v8f acc[4] = {};
    for (int kt = 0; kt < D_IN / 32; ++kt) {
        v16bf a = load_a_f32(X + (size_t)row * D_IN + kt * 32, hi);
#pragma unroll
        for (int u = 0; u < 4; ++u) {
            int nt = ng * 4 + u;
            v16bf b = *(const v16bf*)(const void*)
                      &Wp[((size_t)(kt * 64 + nt)) * 512 + lane * 16];
            acc[u] = __builtin_amdgcn_wmma_f32_16x16x32_bf16(false, a, false, b,
                                                             (short)0, acc[u], false, false);
        }
    }
#pragma unroll
    for (int u = 0; u < 4; ++u)
#pragma unroll
        for (int j = 0; j < 8; ++j) {                  // D: row j+8*hi, col r
            int m = mt * 16 + j + hi * 8;
            int n = (ng * 4 + u) * 16 + r;
            xh[(size_t)m * D_H + n] = f2bfu(acc[u][j] + bh[n]);
        }
}

// ---------------------------------------------------------------------------
__global__ void k_init(const float* __restrict__ hidden, float* __restrict__ hbuf0,
                       unsigned* __restrict__ cnt) {
    int i = blockIdx.x * 256 + threadIdx.x;
    if (i < BATCH * D_H) hbuf0[i] = hidden[i];
    if (i == 0) *cnt = 0u;
}

// ---------------------------------------------------------------------------
// Persistent scan: 16 blocks x 8 waves = 128 waves, 2 N-tiles per wave
// (4 x 64 tile grid of the [64 x 1024] step-GEMM). All waves of a block share
// the row-tile mt -> stage 16x1024 of h in LDS (WMMA-A layout, 32 KB) with
// wide b128 loads, then 2x32 bf16 WMMAs per wave per step.
// ---------------------------------------------------------------------------
#define SCAN_BLOCKS 16

__global__ void k_scan(const unsigned short* __restrict__ xh,
                       const unsigned short* __restrict__ Whh_p,
                       float* __restrict__ hbuf,         // 2 * 64*1024 fp32
                       unsigned short* __restrict__ hs,  // [T*B, D_H] bf16
                       unsigned* __restrict__ cnt) {
    __shared__ unsigned short aLDS[32 * 32 * 16];       // 32 KB packed A tiles
    const int tid  = threadIdx.x;
    const int lane = tid & 31, wave = tid >> 5;
    const int r = lane & 15, hi = lane >> 4;
    const int id  = blockIdx.x * 8 + wave;              // 0..127
    const int mt  = id >> 5;                            // block-uniform
    const int nt0 = (id & 31) * 2;                      // two adjacent N-tiles

    float* buf0 = hbuf;
    float* buf1 = hbuf + BATCH * D_H;

    for (int t = 0; t < T_STEPS; ++t) {
        const float* hin = (t & 1) ? buf1 : buf0;
        float* hout      = (t & 1) ? buf0 : buf1;

        // stage rows [mt*16, mt*16+16) of h into LDS in WMMA-A layout
        for (int d = tid; d < 32 * 32; d += 256) {      // (kt, lane) pairs
            int ln = d & 31, kt = d >> 5;
            int m  = mt * 16 + (ln & 15);
            v16bf frag = load_a_f32(hin + (size_t)m * D_H + kt * 32, ln >> 4);
            *(v16bf*)(void*)&aLDS[d * 16] = frag;       // 32B -> 2x ds_store_b128
        }
        __syncthreads();

        v8f acc0 = {}, acc1 = {};
        for (int kt = 0; kt < D_H / 32; ++kt) {
            v16bf a  = *(const v16bf*)(const void*)&aLDS[(kt * 32 + lane) * 16];
            v16bf b0 = *(const v16bf*)(const void*)
                       &Whh_p[((size_t)(kt * 64 + nt0)) * 512 + lane * 16];
            v16bf b1 = *(const v16bf*)(const void*)
                       &Whh_p[((size_t)(kt * 64 + nt0 + 1)) * 512 + lane * 16];
            if (kt + 1 < D_H / 32)                      // L2 -> WGP$ prefetch
                __builtin_prefetch(&Whh_p[((size_t)((kt + 1) * 64 + nt0)) * 512 + lane * 16], 0, 0);
            acc0 = __builtin_amdgcn_wmma_f32_16x16x32_bf16(false, a, false, b0,
                                                           (short)0, acc0, false, false);
            acc1 = __builtin_amdgcn_wmma_f32_16x16x32_bf16(false, a, false, b1,
                                                           (short)0, acc1, false, false);
        }

#pragma unroll
        for (int j = 0; j < 8; ++j) {
            int m = mt * 16 + j + hi * 8;
            size_t tb = (size_t)t * BATCH + m;
            int n0 = nt0 * 16 + r;
            int n1 = (nt0 + 1) * 16 + r;
            float h0 = tanhf(acc0[j] + bfu2f(xh[tb * D_H + n0]));
            float h1 = tanhf(acc1[j] + bfu2f(xh[tb * D_H + n1]));
            hout[m * D_H + n0] = h0;
            hout[m * D_H + n1] = h1;
            hs[tb * D_H + n0] = f2bfu(h0);
            hs[tb * D_H + n1] = f2bfu(h1);
        }

        // device-wide barrier: monotonic counter reaches (t+1)*SCAN_BLOCKS
        __syncthreads();
        __threadfence();
        if (tid == 0) {
            __hip_atomic_fetch_add(cnt, 1u, __ATOMIC_RELEASE, __HIP_MEMORY_SCOPE_AGENT);
            unsigned target = (unsigned)(t + 1) * SCAN_BLOCKS;
            while (__hip_atomic_load(cnt, __ATOMIC_ACQUIRE, __HIP_MEMORY_SCOPE_AGENT) < target)
                __builtin_amdgcn_s_sleep(2);
        }
        __syncthreads();
    }
}

// ---------------------------------------------------------------------------
// out = hs @ W_hq + b_q -> fp32 [T*B, D_OUT].  4 N-tiles per wave.
// ---------------------------------------------------------------------------
__global__ void k_gemm_out(const unsigned short* __restrict__ hs,
                           const unsigned short* __restrict__ Wp,
                           const float* __restrict__ bq,
                           float* __restrict__ out) {
    const int tid  = threadIdx.x;
    const int lane = tid & 31, wave = tid >> 5;
    const int r = lane & 15, hi = lane >> 4;
    const int job = blockIdx.x * 8 + wave;              // 2048 * 8 jobs
    const int mt  = job >> 3;                           // 0..2047
    const int ng  = job & 7;                            // group of 4 N-tiles
    const int row = mt * 16 + r;

    v8f acc[4] = {};
    for (int kt = 0; kt < D_H / 32; ++kt) {
        v16bf a = load_a_bf16(hs + (size_t)row * D_H + kt * 32, hi);
#pragma unroll
        for (int u = 0; u < 4; ++u) {
            int nt = ng * 4 + u;
            v16bf b = *(const v16bf*)(const void*)
                      &Wp[((size_t)(kt * 32 + nt)) * 512 + lane * 16];
            acc[u] = __builtin_amdgcn_wmma_f32_16x16x32_bf16(false, a, false, b,
                                                             (short)0, acc[u], false, false);
        }
    }
#pragma unroll
    for (int u = 0; u < 4; ++u)
#pragma unroll
        for (int j = 0; j < 8; ++j) {
            int m = mt * 16 + j + hi * 8;
            int n = (ng * 4 + u) * 16 + r;
            out[(size_t)m * D_OUT + n] = acc[u][j] + bq[n];
        }
}

// ---------------------------------------------------------------------------
__global__ void k_final(const float* __restrict__ hbuf0, float* __restrict__ dst) {
    int i = blockIdx.x * 256 + threadIdx.x;
    if (i < BATCH * D_H) dst[i] = hbuf0[i];
}

// ---------------------------------------------------------------------------
extern "C" void kernel_launch(void* const* d_in, const int* in_sizes, int n_in,
                              void* d_out, int out_size, void* d_ws, size_t ws_size,
                              hipStream_t stream) {
    (void)in_sizes; (void)n_in; (void)out_size; (void)ws_size;

    const float* X      = (const float*)d_in[0];   // [T,B,D_IN]
    const float* hidden = (const float*)d_in[1];   // [B,D_H]
    const float* W_xh   = (const float*)d_in[2];   // [D_IN,D_H]
    const float* W_hh   = (const float*)d_in[3];   // [D_H,D_H]
    const float* b_h    = (const float*)d_in[4];   // [D_H]
    const float* W_hq   = (const float*)d_in[5];   // [D_H,D_OUT]
    const float* b_q    = (const float*)d_in[6];   // [D_OUT]
    float* out = (float*)d_out;                    // [T*B*D_OUT] ++ [B*D_H]

    // workspace carve (all offsets 256B aligned); total ~132.5 MiB
    char* ws = (char*)d_ws;
    unsigned short* Wxh_p = (unsigned short*)(ws);                       // 1 MiB
    unsigned short* Whh_p = (unsigned short*)(ws + (1u << 20));          // 2 MiB
    unsigned short* Whq_p = (unsigned short*)(ws + (3u << 20));          // 1 MiB
    unsigned short* xh    = (unsigned short*)(ws + (4u << 20));          // 64 MiB
    unsigned short* hs    = (unsigned short*)(ws + (68ull << 20));       // 64 MiB
    float*          hbuf  = (float*)(ws + (132ull << 20));               // 512 KiB
    unsigned*       cnt   = (unsigned*)(ws + (132ull << 20) + (512u << 10));

    // 1) pack weights to bf16 WMMA-B layout
    k_pack<<<(D_IN * D_H) / 256, 256, 0, stream>>>(W_xh, Wxh_p, D_H);
    k_pack<<<(D_H * D_H) / 256, 256, 0, stream>>>(W_hh, Whh_p, D_H);
    k_pack<<<(D_H * D_OUT) / 256, 256, 0, stream>>>(W_hq, Whq_p, D_OUT);

    // 2) xh = X @ W_xh + b_h   (2048*16 wave-jobs, 8 waves per block)
    k_gemm_xh<<<4096, 256, 0, stream>>>(X, Wxh_p, b_h, xh);

    // 3) init h buffer + barrier counter (reset every call -> deterministic)
    k_init<<<(BATCH * D_H + 255) / 256, 256, 0, stream>>>(hidden, hbuf, cnt);

    // 4) sequential scan, persistent cooperative kernel
    k_scan<<<SCAN_BLOCKS, 256, 0, stream>>>(xh, Whh_p, hbuf, hs, cnt);

    // 5) out = hs @ W_hq + b_q  (2048*8 wave-jobs)
    k_gemm_out<<<2048, 256, 0, stream>>>(hs, Whq_p, b_q, out);

    // 6) final hidden -> tail of d_out (t=511 writes buf0)
    k_final<<<(BATCH * D_H + 255) / 256, 256, 0, stream>>>(
        hbuf, out + (size_t)T_STEPS * BATCH * D_OUT);
}
